// RNNDecoderWithAttention_62947040690533
// MI455X (gfx1250) — compile-verified, matrix operand order checked by reading, not run on hardware
//
#include <hip/hip_runtime.h>
#include <hip/hip_bf16.h>
#include <cstdint>
#include <cstddef>

// ---------------------------------------------------------------------------
// Types / helpers
// ---------------------------------------------------------------------------
typedef __attribute__((ext_vector_type(16))) __bf16 v16bf;
typedef __attribute__((ext_vector_type(8)))  float  v8f;

#define DEV_INLINE __device__ __forceinline__

DEV_INLINE unsigned short f2bf(float f) {
  unsigned int u = __float_as_uint(f);
  unsigned int r = (u + 0x7FFFu + ((u >> 16) & 1u)) >> 16;  // round-to-nearest-even
  return (unsigned short)r;
}
DEV_INLINE float bf2f(unsigned short u) {
  return __uint_as_float(((unsigned int)u) << 16);
}
DEV_INLINE float sigmoidf_(float x) { return 1.0f / (1.0f + expf(-x)); }

struct alignas(16) U128 { unsigned int x, y, z, w; };
struct U256 { U128 lo, hi; };

// Load one lane's 16-element bf16 fragment: elements [0..8) from p,
// elements [8..16) from p+16 (matches 16-bit A/B 16x32 VGPR layout:
// lane half 0 holds K {0..7,16..23}, half 1 holds K {8..15,24..31}).
DEV_INLINE v16bf load_frag(const unsigned short* p) {
  U256 t;
  t.lo = *reinterpret_cast<const U128*>(p);
  t.hi = *reinterpret_cast<const U128*>(p + 16);
  return __builtin_bit_cast(v16bf, t);
}

DEV_INLINE v8f wmma_bf16(v16bf a, v16bf b, v8f c) {
  return __builtin_amdgcn_wmma_f32_16x16x32_bf16(false, a, false, b, (short)0, c,
                                                 false, false);
}

DEV_INLINE void store_tile(float* __restrict__ C, long ldc, int mbase, int n,
                           const float* __restrict__ bias, v8f acc) {
  float bz = bias ? bias[n] : 0.0f;
#pragma unroll
  for (int r = 0; r < 8; ++r) C[(size_t)(mbase + r) * ldc + n] = acc[r] + bz;
}

// ---------------------------------------------------------------------------
// WMMA GEMM: C[M, NT*16*gridDim.x] = A[M,K](bf16, lda) * Bt[N,K]^T(bf16, ldb) + bias
// One wave per 16 x (NT*16) tile. NT is compile-time: no branches in the
// K-loop, EXEC uniform (required for WMMA), scalar accumulators.
// Distance-1 software pipeline: next iteration's fragments are loaded before
// the current WMMAs so loads stay in flight during the matrix ops.
// M multiple of 16, K multiple of 32; every launch covers full 16-wide tiles.
// ---------------------------------------------------------------------------
template <int NT>
__global__ __launch_bounds__(32) void gemm_bf16_wmma(
    const unsigned short* __restrict__ A, int lda,
    const unsigned short* __restrict__ Bt, int ldb,
    const float* __restrict__ bias,
    float* __restrict__ C, long ldc, int K) {
  const int lane  = threadIdx.x;
  const int m0    = blockIdx.y << 4;
  const int n0    = blockIdx.x * (NT * 16);
  const int row   = lane & 15;
  const int khalf = (lane >> 4) << 3;  // 0 or 8

  const unsigned short* pa  = A + (size_t)(m0 + row) * lda + khalf;
  const unsigned short* pb0 = Bt + (size_t)(n0 + row) * ldb + khalf;
  const unsigned short* pb1 = pb0 + (size_t)16 * ldb;
  const unsigned short* pb2 = pb0 + (size_t)32 * ldb;
  const unsigned short* pb3 = pb0 + (size_t)48 * ldb;

  v8f acc0 = {0, 0, 0, 0, 0, 0, 0, 0};
  v8f acc1 = acc0, acc2 = acc0, acc3 = acc0;

  // ---- prologue: load fragments for k = 0
  v16bf a0 = load_frag(pa);
  pa += 32;
  v16bf b0 = load_frag(pb0);
  pb0 += 32;
  v16bf b1 = b0, b2 = b0, b3 = b0;
  if (NT > 1) { b1 = load_frag(pb1); pb1 += 32; }
  if (NT > 2) { b2 = load_frag(pb2); pb2 += 32; }
  if (NT > 3) { b3 = load_frag(pb3); pb3 += 32; }

  // ---- steady state: prefetch k+32, compute k
  for (int k0 = 32; k0 < K; k0 += 32) {
    v16bf an = load_frag(pa);
    pa += 32;
    v16bf bn0 = load_frag(pb0);
    pb0 += 32;
    v16bf bn1 = bn0, bn2 = bn0, bn3 = bn0;
    if (NT > 1) { bn1 = load_frag(pb1); pb1 += 32; }
    if (NT > 2) { bn2 = load_frag(pb2); pb2 += 32; }
    if (NT > 3) { bn3 = load_frag(pb3); pb3 += 32; }

    acc0 = wmma_bf16(a0, b0, acc0);
    if (NT > 1) acc1 = wmma_bf16(a0, b1, acc1);
    if (NT > 2) acc2 = wmma_bf16(a0, b2, acc2);
    if (NT > 3) acc3 = wmma_bf16(a0, b3, acc3);

    a0 = an;
    b0 = bn0;
    if (NT > 1) b1 = bn1;
    if (NT > 2) b2 = bn2;
    if (NT > 3) b3 = bn3;
  }

  // ---- epilogue: last k-chunk
  acc0 = wmma_bf16(a0, b0, acc0);
  if (NT > 1) acc1 = wmma_bf16(a0, b1, acc1);
  if (NT > 2) acc2 = wmma_bf16(a0, b2, acc2);
  if (NT > 3) acc3 = wmma_bf16(a0, b3, acc3);

  const int mbase = m0 + ((lane >> 4) << 3);
  store_tile(C, ldc, mbase, n0 + row, bias, acc0);
  if (NT > 1) store_tile(C, ldc, mbase, n0 + 16 + row, bias, acc1);
  if (NT > 2) store_tile(C, ldc, mbase, n0 + 32 + row, bias, acc2);
  if (NT > 3) store_tile(C, ldc, mbase, n0 + 48 + row, bias, acc3);
}

// ---------------------------------------------------------------------------
// Transpose + f32->bf16 convert: in[K,N] -> out[n*ld_out + k_off + k]
// ---------------------------------------------------------------------------
__global__ void transpose_f32_to_bf16(const float* __restrict__ in,
                                      unsigned short* __restrict__ out,
                                      int K, int N, int ld_out, int k_off) {
  __shared__ float tile[32][33];
  int kb = blockIdx.x * 32, nb = blockIdx.y * 32;
  int tx = threadIdx.x, ty = threadIdx.y;  // 32 x 8
#pragma unroll
  for (int j = 0; j < 4; ++j) {
    int k = kb + ty + j * 8, n = nb + tx;
    if (k < K && n < N) tile[ty + j * 8][tx] = in[(size_t)k * N + n];
  }
  __syncthreads();
#pragma unroll
  for (int j = 0; j < 4; ++j) {
    int n = nb + ty + j * 8, k = kb + tx;
    if (n < N && k < K)
      out[(size_t)n * ld_out + k_off + k] = f2bf(tile[tx][ty + j * 8]);
  }
}

__global__ void cvt_f32_bf16(const float* __restrict__ in,
                             unsigned short* __restrict__ out, long n) {
  long i = (long)blockIdx.x * blockDim.x + threadIdx.x;
  long stride = (long)gridDim.x * blockDim.x;
  for (; i < n; i += stride) out[i] = f2bf(in[i]);
}

// mean over P=196: enc[64,196,2048] -> mean_bf[64,2048] (bf16)
__global__ void mean_kernel(const float* __restrict__ enc,
                            unsigned short* __restrict__ mean_bf) {
  int b = blockIdx.x, e = blockIdx.y * 256 + threadIdx.x;
  const float* p = enc + (size_t)b * 196 * 2048 + e;
  float s = 0.0f;
  for (int i = 0; i < 196; ++i) s += p[(size_t)i * 2048];
  mean_bf[b * 2048 + e] = f2bf(s * (1.0f / 196.0f));
}

// copy f32 h0[64,2048] into the h-section (cols 2560..4607) of xh_bf[64,4608]
__global__ void h_to_xh(const float* __restrict__ h,
                        unsigned short* __restrict__ xh) {
  int i = blockIdx.x * 256 + threadIdx.x;  // 0..131071
  int b = i >> 11, d = i & 2047;
  xh[b * 4608 + 2560 + d] = f2bf(h[i]);
}

__global__ void add_bias(const float* __restrict__ a,
                         const float* __restrict__ b,
                         float* __restrict__ o, int n) {
  int i = blockIdx.x * 256 + threadIdx.x;
  if (i < n) o[i] = a[i] + b[i];
}

// additive attention score + softmax over P=196, one block per batch row
__global__ void att_softmax(const unsigned short* __restrict__ enc_proj,  // [64,196,512] bf16
                            const float* __restrict__ hid_proj,           // [64,512]
                            const float* __restrict__ full_W,             // [512]
                            const float* __restrict__ full_b,             // [1]
                            float* __restrict__ att_w,                    // [64,196]
                            float* __restrict__ att_out) {                // d_out att + t*196
  int b = blockIdx.x, tid = threadIdx.x;
  __shared__ float s_h[512];
  __shared__ float s_w[512];
  __shared__ float red[256];
  for (int a = tid; a < 512; a += 256) {
    s_h[a] = hid_proj[b * 512 + a];
    s_w[a] = full_W[a];
  }
  __syncthreads();

  float sc = -1e30f;
  if (tid < 196) {
    const unsigned short* ep = enc_proj + (size_t)(b * 196 + tid) * 512;
    float s = full_b[0];
    for (int a = 0; a < 512; ++a) {
      float v = bf2f(ep[a]) + s_h[a];
      s += fmaxf(v, 0.0f) * s_w[a];
    }
    sc = s;
  }
  red[tid] = sc;
  __syncthreads();
  for (int o = 128; o > 0; o >>= 1) {
    if (tid < o) red[tid] = fmaxf(red[tid], red[tid + o]);
    __syncthreads();
  }
  float mx = red[0];
  __syncthreads();
  float ex = (tid < 196) ? expf(sc - mx) : 0.0f;
  red[tid] = ex;
  __syncthreads();
  for (int o = 128; o > 0; o >>= 1) {
    if (tid < o) red[tid] += red[tid + o];
    __syncthreads();
  }
  float inv = 1.0f / red[0];
  if (tid < 196) {
    float w = ex * inv;
    att_w[b * 196 + tid] = w;
    att_out[(size_t)b * (25 * 196) + tid] = w;
  }
}

// ctx[b,e] = sigmoid(fbeta[b,e]) * sum_p enc_bf[b,p,e]*att_w[b,p]
// written as bf16 into ctx-section (cols 512..2559) of xh_bf
__global__ void ctx_gate(const unsigned short* __restrict__ enc_bf,  // [64,196,2048]
                         const float* __restrict__ att_w,            // [64,196]
                         const float* __restrict__ fbeta,            // [64,2048]
                         unsigned short* __restrict__ xh) {
  int b = blockIdx.x, e = blockIdx.y * 256 + threadIdx.x;
  __shared__ float s_aw[196];
  if (threadIdx.x < 196) s_aw[threadIdx.x] = att_w[b * 196 + threadIdx.x];
  __syncthreads();
  const unsigned short* p = enc_bf + (size_t)b * 196 * 2048 + e;
  float s = 0.0f;
  for (int i = 0; i < 196; ++i) s += bf2f(p[(size_t)i * 2048]) * s_aw[i];
  float g = sigmoidf_(fbeta[b * 2048 + e]);
  xh[b * 4608 + 512 + e] = f2bf(s * g);
}

// xh[b, 0..511] = bf16(emb[caption_gt[b,t]])
__global__ void gather_emb(const float* __restrict__ emb,
                           const int* __restrict__ cap, int t,
                           unsigned short* __restrict__ xh) {
  int b = blockIdx.x, j = blockIdx.y * 256 + threadIdx.x;  // j < 512
  int tok = cap[b * 25 + t];
  xh[b * 4608 + j] = f2bf(emb[(size_t)tok * 512 + j]);
}

// LSTM pointwise: gates[64,8192] (i,f,g,o), c[64,2048]; writes c and h->xh(bf16)
__global__ void lstm_cell(const float* __restrict__ gates,
                          float* __restrict__ c,
                          unsigned short* __restrict__ xh) {
  int i = blockIdx.x * 256 + threadIdx.x;  // 0..131071
  int b = i >> 11, d = i & 2047;
  const float* g = gates + (size_t)b * 8192;
  float ig = sigmoidf_(g[d]);
  float fg = sigmoidf_(g[2048 + d]);
  float gg = tanhf(g[4096 + d]);
  float og = sigmoidf_(g[6144 + d]);
  float cn = fg * c[i] + ig * gg;
  float hn = og * tanhf(cn);
  c[i] = cn;
  xh[b * 4608 + 2560 + d] = f2bf(hn);
}

__global__ void cap_to_out(const int* __restrict__ cap,
                           float* __restrict__ out) {
  int i = blockIdx.x * 256 + threadIdx.x;
  if (i < 1600) out[i] = (float)cap[i];
}

// ---------------------------------------------------------------------------
// Host orchestration
// ---------------------------------------------------------------------------
extern "C" void kernel_launch(void* const* d_in, const int* in_sizes, int n_in,
                              void* d_out, int out_size, void* d_ws, size_t ws_size,
                              hipStream_t stream) {
  (void)in_sizes; (void)n_in; (void)out_size; (void)ws_size;

  const float* enc        = (const float*)d_in[0];
  const int*   cap        = (const int*)d_in[1];
  const float* enc_att_W  = (const float*)d_in[2];
  const float* enc_att_b  = (const float*)d_in[3];
  const float* hid_att_W  = (const float*)d_in[4];
  const float* hid_att_b  = (const float*)d_in[5];
  const float* full_att_W = (const float*)d_in[6];
  const float* full_att_b = (const float*)d_in[7];
  const float* emb        = (const float*)d_in[8];
  const float* W_ih       = (const float*)d_in[9];
  const float* b_ih       = (const float*)d_in[10];
  const float* W_hh       = (const float*)d_in[11];
  const float* b_hh       = (const float*)d_in[12];
  const float* init_h_W   = (const float*)d_in[13];
  const float* init_h_b   = (const float*)d_in[14];
  const float* init_c_W   = (const float*)d_in[15];
  const float* init_c_b   = (const float*)d_in[16];
  const float* f_beta_W   = (const float*)d_in[17];
  const float* f_beta_b   = (const float*)d_in[18];
  const float* fc_W       = (const float*)d_in[19];
  const float* fc_b       = (const float*)d_in[20];

  char* ws = (char*)d_ws;
  size_t off = 0;
  auto alloc = [&](size_t bytes) -> void* {
    off = (off + 255) & ~(size_t)255;
    void* p = ws + off;
    off += bytes;
    return p;
  };

  // bf16 transposed weights (persist across the whole launch)
  unsigned short* enc_att_Wt  = (unsigned short*)alloc(512ull * 2048 * 2);
  unsigned short* hid_att_Wt  = (unsigned short*)alloc(512ull * 2048 * 2);
  unsigned short* f_beta_Wt   = (unsigned short*)alloc(2048ull * 2048 * 2);
  unsigned short* W_full_t    = (unsigned short*)alloc(8192ull * 4608 * 2);
  unsigned short* fc_Wt       = (unsigned short*)alloc(10000ull * 2048 * 2);
  unsigned short* init_h_Wt   = (unsigned short*)alloc(2048ull * 2048 * 2);
  unsigned short* init_c_Wt   = (unsigned short*)alloc(2048ull * 2048 * 2);
  unsigned short* enc_bf      = (unsigned short*)alloc(64ull * 196 * 2048 * 2);
  unsigned short* enc_proj_bf = (unsigned short*)alloc(64ull * 196 * 512 * 2);
  unsigned short* mean_bf     = (unsigned short*)alloc(64ull * 2048 * 2);
  unsigned short* xh_bf       = (unsigned short*)alloc(64ull * 4608 * 2);  // [emb|ctx|h]
  float* enc_proj_f32 = (float*)alloc(64ull * 196 * 512 * 4);
  float* c_st         = (float*)alloc(64ull * 2048 * 4);
  float* hid_proj     = (float*)alloc(64ull * 512 * 4);
  float* fbeta_out    = (float*)alloc(64ull * 2048 * 4);
  float* att_w        = (float*)alloc(64ull * 196 * 4);
  float* gates        = (float*)alloc(64ull * 8192 * 4);
  float* bias_c       = (float*)alloc(8192ull * 4);
  float* tmpH         = (float*)alloc(64ull * 2048 * 4);

  dim3 tb(32, 8);
  // one-time: weight transposes + conversions
  transpose_f32_to_bf16<<<dim3(64, 16), tb, 0, stream>>>(enc_att_W, enc_att_Wt, 2048, 512, 2048, 0);
  transpose_f32_to_bf16<<<dim3(64, 16), tb, 0, stream>>>(hid_att_W, hid_att_Wt, 2048, 512, 2048, 0);
  transpose_f32_to_bf16<<<dim3(64, 64), tb, 0, stream>>>(f_beta_W, f_beta_Wt, 2048, 2048, 2048, 0);
  transpose_f32_to_bf16<<<dim3(80, 256), tb, 0, stream>>>(W_ih, W_full_t, 2560, 8192, 4608, 0);
  transpose_f32_to_bf16<<<dim3(64, 256), tb, 0, stream>>>(W_hh, W_full_t, 2048, 8192, 4608, 2560);
  transpose_f32_to_bf16<<<dim3(64, 313), tb, 0, stream>>>(fc_W, fc_Wt, 2048, 10000, 2048, 0);
  transpose_f32_to_bf16<<<dim3(64, 64), tb, 0, stream>>>(init_h_W, init_h_Wt, 2048, 2048, 2048, 0);
  transpose_f32_to_bf16<<<dim3(64, 64), tb, 0, stream>>>(init_c_W, init_c_Wt, 2048, 2048, 2048, 0);

  cvt_f32_bf16<<<2048, 256, 0, stream>>>(enc, enc_bf, 64ll * 196 * 2048);
  mean_kernel<<<dim3(64, 8), 256, 0, stream>>>(enc, mean_bf);
  add_bias<<<32, 256, 0, stream>>>(b_ih, b_hh, bias_c, 8192);

  // init h, c : mean @ init_{h,c}_W + b
  gemm_bf16_wmma<4><<<dim3(32, 4), 32, 0, stream>>>(mean_bf, 2048, init_h_Wt, 2048, init_h_b, tmpH, 2048, 2048);
  gemm_bf16_wmma<4><<<dim3(32, 4), 32, 0, stream>>>(mean_bf, 2048, init_c_Wt, 2048, init_c_b, c_st, 2048, 2048);
  h_to_xh<<<512, 256, 0, stream>>>(tmpH, xh_bf);

  // one-time: enc_proj = enc @ enc_att_W + b  (M = 64*196 = 12544)
  gemm_bf16_wmma<4><<<dim3(8, 784), 32, 0, stream>>>(enc_bf, 2048, enc_att_Wt, 2048, enc_att_b, enc_proj_f32, 512, 2048);
  cvt_f32_bf16<<<2048, 256, 0, stream>>>(enc_proj_f32, enc_proj_bf, 64ll * 196 * 512);

  float* out_pred = (float*)d_out;                    // [64,25,10000]
  float* out_cap  = out_pred + 64ll * 25 * 10000;     // [64,25] as float
  float* out_att  = out_cap + 64 * 25;                // [64,25,196]
  cap_to_out<<<7, 256, 0, stream>>>(cap, out_cap);

  for (int t = 0; t < 25; ++t) {
    // hid_proj = h @ hid_att_W + b
    gemm_bf16_wmma<4><<<dim3(8, 4), 32, 0, stream>>>(xh_bf + 2560, 4608, hid_att_Wt, 2048, hid_att_b, hid_proj, 512, 2048);
    // fbeta_out = h @ f_beta_W + b
    gemm_bf16_wmma<4><<<dim3(32, 4), 32, 0, stream>>>(xh_bf + 2560, 4608, f_beta_Wt, 2048, f_beta_b, fbeta_out, 2048, 2048);
    // attention scores + softmax (also writes att weights output)
    att_softmax<<<64, 256, 0, stream>>>(enc_proj_bf, hid_proj, full_att_W, full_att_b, att_w, out_att + (size_t)t * 196);
    // x = [emb_t | gated ctx] (bf16 into xh)
    gather_emb<<<dim3(64, 2), 256, 0, stream>>>(emb, cap, t, xh_bf);
    ctx_gate<<<dim3(64, 8), 256, 0, stream>>>(enc_bf, att_w, fbeta_out, xh_bf);
    // gates = [x|h] @ [W_ih;W_hh] + (b_ih + b_hh)
    gemm_bf16_wmma<4><<<dim3(128, 4), 32, 0, stream>>>(xh_bf, 4608, W_full_t, 4608, bias_c, gates, 8192, 4608);
    // LSTM pointwise -> c, h(bf16 into xh)
    lstm_cell<<<512, 256, 0, stream>>>(gates, c_st, xh_bf);
    // logits -> d_out[b, t, :]  (9984 cols with NT=4, last 16 with NT=1)
    gemm_bf16_wmma<4><<<dim3(156, 4), 32, 0, stream>>>(xh_bf + 2560, 4608, fc_Wt, 2048, fc_b, out_pred + (size_t)t * 10000, 25ll * 10000, 2048);
    gemm_bf16_wmma<1><<<dim3(1, 4), 32, 0, stream>>>(xh_bf + 2560, 4608, fc_Wt + 9984ull * 2048, 2048, fc_b + 9984, out_pred + (size_t)t * 10000 + 9984, 25ll * 10000, 2048);
  }
}
